// TorMamba_27882927686408
// MI455X (gfx1250) — matrix operand order repaired and checked by reading, not compiled, not run on hardware
//
#include <hip/hip_runtime.h>
#include <hip/hip_bf16.h>

// ---------------- model constants ----------------
#define D_MODEL 192
#define D_INNER 384
#define D_STATE 16
#define D_CONV  4
#define DT_RANK 12
#define DEPTH   4
#define PSTRIDE 4
#define N_PATCH 400
#define NCLS    1000
#define BATCH   32
#define L_SEQ   (N_PATCH + 1)          // 401
#define M_TOK   (BATCH * L_SEQ)        // 12832
#define EPSF    1e-5f

typedef __attribute__((ext_vector_type(16))) _Float16 v16h;
typedef __attribute__((ext_vector_type(8)))  _Float16 v8h;
typedef __attribute__((ext_vector_type(8)))  float    v8f;

#if __has_builtin(__builtin_amdgcn_global_load_async_to_lds_b128) && \
    __has_builtin(__builtin_amdgcn_s_wait_asynccnt)
#define HAVE_ASYNC_LDS 1
// ROCm 7.2 declares the async-LDS builtin with typed v4i pointers:
//   arg0: int __vector(4) AS(1)*   (global source)
//   arg1: int __vector(4) AS(3)*   (LDS destination)
typedef int v4i_async __attribute__((vector_size(16)));
typedef __attribute__((address_space(1))) v4i_async* async_gptr_t;
typedef __attribute__((address_space(3))) v4i_async* async_lptr_t;
#else
#define HAVE_ASYNC_LDS 0
#endif

__device__ __forceinline__ float silu_f(float x) {
    return x / (1.0f + __expf(-x));
}

// ---------------- patch embed + pos embed + cls ----------------
__global__ void patch_embed_kernel(const float* __restrict__ imgs,
                                   const float* __restrict__ patch_w,
                                   const float* __restrict__ patch_b,
                                   const float* __restrict__ cls_token,
                                   const float* __restrict__ pos_embed,
                                   float* __restrict__ x) {
    int idx = blockIdx.x * blockDim.x + threadIdx.x;
    if (idx >= M_TOK * D_MODEL) return;
    int token = idx / D_MODEL;
    int c     = idx % D_MODEL;
    int b     = token / L_SEQ;
    int p     = token % L_SEQ;
    float v;
    if (p == N_PATCH) {
        v = cls_token[c] + pos_embed[N_PATCH * D_MODEL + c];
    } else {
        v = patch_b[c] + pos_embed[p * D_MODEL + c];
        const float* im = imgs + (size_t)b * (N_PATCH * PSTRIDE) + p * PSTRIDE;
        #pragma unroll
        for (int s = 0; s < PSTRIDE; ++s)
            v += im[s] * patch_w[s * D_MODEL + c];
    }
    x[idx] = v;
}

// ---------------- residual add + RMSNorm -> f16 activations ----------------
// one wave (32 lanes) per token, 6 channels per lane (192 = 6*32)
__global__ void rmsnorm_kernel(const float* __restrict__ x,
                               float* __restrict__ resid,
                               _Float16* __restrict__ h16,
                               const float* __restrict__ w,
                               int is_first) {
    int wv   = threadIdx.x >> 5;
    int lane = threadIdx.x & 31;
    int token = blockIdx.x * (blockDim.x >> 5) + wv;
    if (token >= M_TOK) return;
    size_t base = (size_t)token * D_MODEL;
    float vals[6];
    float ss = 0.0f;
    #pragma unroll
    for (int i = 0; i < 6; ++i) {
        int c = lane + 32 * i;
        float xv = x[base + c];
        float rv = is_first ? xv : (xv + resid[base + c]);
        resid[base + c] = rv;
        vals[i] = rv;
        ss += rv * rv;
    }
    #pragma unroll
    for (int off = 16; off > 0; off >>= 1)
        ss += __shfl_xor(ss, off);
    float inv = rsqrtf(ss * (1.0f / D_MODEL) + EPSF);
    #pragma unroll
    for (int i = 0; i < 6; ++i) {
        int c = lane + 32 * i;
        h16[base + c] = (_Float16)(vals[i] * inv * w[c]);
    }
}

// ---------------- f32 -> f16 weight conversion ----------------
__global__ void cvt_f16_kernel(const float* __restrict__ src,
                               _Float16* __restrict__ dst, int n) {
    int i = blockIdx.x * blockDim.x + threadIdx.x;
    if (i < n) dst[i] = (_Float16)src[i];
}

// ---------------- WMMA GEMM: C[MxN] = A[MxK] * B[KxN], f16 in, f32 out ----
// block = 128 threads = 4 waves; block tile 64(M) x 64(N); K step 32.
// A tile staged via async global->LDS DMA (ASYNCcnt) when available.
__global__ __launch_bounds__(128)
void gemm_f16_wmma_kernel(const _Float16* __restrict__ A,
                          const _Float16* __restrict__ B,
                          float* __restrict__ C,
                          int M, int N, int K) {
    __shared__ _Float16 lA[64 * 32];   // [row][k]
    __shared__ _Float16 lBt[64 * 32];  // transposed: [n][k]
    const int tid  = threadIdx.x;
    const int wv   = tid >> 5;
    const int lane = tid & 31;
    const int bm = blockIdx.y * 64;
    const int bn = blockIdx.x * 64;
    const int rhalf = lane & 15;
    const int hi    = lane >> 4;

    v8f acc[4] = {};

    for (int k0 = 0; k0 < K; k0 += 32) {
        // stage A tile 64x32 (async DMA global->LDS; zero-fill OOB rows)
        #pragma unroll
        for (int i = 0; i < 2; ++i) {
            int chunk = tid + i * 128;          // 256 chunks of 8 halfs
            int row = chunk >> 2, cc = chunk & 3;
            int grow = bm + row;
            _Float16* lp = &lA[row * 32 + cc * 8];
#if HAVE_ASYNC_LDS
            if (grow < M) {
                const _Float16* gp = A + (size_t)grow * K + k0 + cc * 8;
                __builtin_amdgcn_global_load_async_to_lds_b128(
                    (async_gptr_t)(uintptr_t)gp, (async_lptr_t)(uintptr_t)lp, 0, 0);
            } else {
                *(uint4*)lp = make_uint4(0u, 0u, 0u, 0u);
            }
#else
            uint4 v = make_uint4(0u, 0u, 0u, 0u);
            if (grow < M)
                v = *(const uint4*)(A + (size_t)grow * K + k0 + cc * 8);
            *(uint4*)lp = v;
#endif
        }
        // stage B tile 32x64, transposed into lBt[n][k]
        #pragma unroll
        for (int i = 0; i < 2; ++i) {
            int chunk = tid + i * 128;
            int k = chunk >> 3, nc = chunk & 7;
            uint4 v = *(const uint4*)(B + (size_t)(k0 + k) * N + bn + nc * 8);
            _Float16 tmp[8];
            *(uint4*)tmp = v;
            #pragma unroll
            for (int j = 0; j < 8; ++j)
                lBt[(nc * 8 + j) * 32 + k] = tmp[j];
        }
        // speculative prefetch of next A K-slab
        if (k0 + 32 < K) {
            int grow = bm + (tid >> 1);
            if (grow < M)
                __builtin_prefetch(A + (size_t)grow * K + k0 + 32, 0, 0);
        }
#if HAVE_ASYNC_LDS
        __builtin_amdgcn_s_wait_asynccnt(0);
#endif
        __syncthreads();

        // A fragment: lanes 0-15 row=r, K {0..7,16..23}; lanes 16-31 K {8..15,24..31}
        const _Float16* ap = &lA[(wv * 16 + rhalf) * 32 + hi * 8];
        v8h a0 = *(const v8h*)(ap);
        v8h a1 = *(const v8h*)(ap + 16);
        v16h afrag;
        #pragma unroll
        for (int i = 0; i < 8; ++i) { afrag[i] = a0[i]; afrag[i + 8] = a1[i]; }

        #pragma unroll
        for (int t = 0; t < 4; ++t) {
            // B fragment: lanes 0-15 col=n, K {0..7,16..23}; lanes 16-31 K {8..15,24..31}
            const _Float16* bp = &lBt[(t * 16 + rhalf) * 32 + hi * 8];
            v8h b0 = *(const v8h*)(bp);
            v8h b1 = *(const v8h*)(bp + 16);
            v16h bfrag;
            #pragma unroll
            for (int i = 0; i < 8; ++i) { bfrag[i] = b0[i]; bfrag[i + 8] = b1[i]; }
            acc[t] = __builtin_amdgcn_wmma_f32_16x16x32_f16(
                false, afrag, false, bfrag, (short)0, acc[t], false, false);
        }
        __syncthreads();
    }

    // epilogue: C/D layout — VGPR v: lanes 0-15 row=v, lanes 16-31 row=v+8; col=lane&15
    #pragma unroll
    for (int t = 0; t < 4; ++t) {
        #pragma unroll
        for (int v = 0; v < 8; ++v) {
            int row = bm + wv * 16 + hi * 8 + v;
            int col = bn + t * 16 + rhalf;
            if (row < M) C[(size_t)row * N + col] = acc[t][v];
        }
    }
}

// ---------------- causal depthwise conv (k=4) + SiLU ----------------
__global__ void conv_silu_kernel(const float* __restrict__ xz,
                                 const float* __restrict__ conv_w,
                                 const float* __restrict__ conv_b,
                                 float* __restrict__ uc) {
    int idx = blockIdx.x * blockDim.x + threadIdx.x;
    if (idx >= M_TOK * D_INNER) return;
    int token = idx / D_INNER;
    int d     = idx % D_INNER;
    int b     = token / L_SEQ;
    int l     = token % L_SEQ;
    float acc = conv_b[d];
    #pragma unroll
    for (int k = 0; k < D_CONV; ++k) {
        int ll = l + k - (D_CONV - 1);
        if (ll >= 0)
            acc += xz[(size_t)(b * L_SEQ + ll) * (2 * D_INNER) + d] * conv_w[d * D_CONV + k];
    }
    uc[idx] = silu_f(acc);
}

// ---------------- x_proj: dbl = uc @ xp_w (384 x 44) ----------------
__global__ void xproj_kernel(const float* __restrict__ uc,
                             const float* __restrict__ xp_w,
                             float* __restrict__ dbl) {
    int idx = blockIdx.x * blockDim.x + threadIdx.x;
    if (idx >= M_TOK * (DT_RANK + 2 * D_STATE)) return;
    int token = idx / (DT_RANK + 2 * D_STATE);
    int j     = idx % (DT_RANK + 2 * D_STATE);
    const float* ur = uc + (size_t)token * D_INNER;
    float s = 0.0f;
    for (int k = 0; k < D_INNER; ++k)
        s += ur[k] * xp_w[k * (DT_RANK + 2 * D_STATE) + j];
    dbl[idx] = s;
}

// ---------------- dt_proj + softplus ----------------
__global__ void dtproj_kernel(const float* __restrict__ dbl,
                              const float* __restrict__ dt_w,
                              const float* __restrict__ dt_b,
                              float* __restrict__ delta) {
    int idx = blockIdx.x * blockDim.x + threadIdx.x;
    if (idx >= M_TOK * D_INNER) return;
    int token = idx / D_INNER;
    int d     = idx % D_INNER;
    const float* dr = dbl + (size_t)token * (DT_RANK + 2 * D_STATE);
    float s = dt_b[d];
    #pragma unroll
    for (int r = 0; r < DT_RANK; ++r)
        s += dr[r] * dt_w[r * D_INNER + d];
    delta[idx] = (s > 20.0f) ? s : log1pf(__expf(s));
}

// ---------------- selective scan: h[16] in registers, serial over L ----------
// one thread per (b, d); y written into the (no longer needed) u half of xz
__global__ void scan_kernel(const float* __restrict__ delta,
                            const float* __restrict__ uc,
                            const float* __restrict__ dbl,
                            const float* __restrict__ A_log,
                            float* __restrict__ xz /* y -> first half */) {
    int gid = blockIdx.x * blockDim.x + threadIdx.x;
    if (gid >= BATCH * D_INNER) return;
    int b = gid / D_INNER;
    int d = gid % D_INNER;
    float A[D_STATE], h[D_STATE];
    #pragma unroll
    for (int s = 0; s < D_STATE; ++s) {
        A[s] = -__expf(A_log[d * D_STATE + s]);
        h[s] = 0.0f;
    }
    for (int l = 0; l < L_SEQ; ++l) {
        size_t base = (size_t)(b * L_SEQ + l);
        float dv = delta[base * D_INNER + d];
        float uv = uc[base * D_INNER + d];
        float du = dv * uv;
        const float* bc = dbl + base * (DT_RANK + 2 * D_STATE);
        float y = 0.0f;
        #pragma unroll
        for (int s = 0; s < D_STATE; ++s) {
            float Bm = bc[DT_RANK + s];
            float Cm = bc[DT_RANK + D_STATE + s];
            h[s] = __expf(dv * A[s]) * h[s] + du * Bm;
            y += h[s] * Cm;
        }
        xz[base * (2 * D_INNER) + d] = y;
    }
}

// ---------------- gate: (y + uc*D) * silu(z) -> f16 ----------------
__global__ void gate_kernel(const float* __restrict__ xz,
                            const float* __restrict__ uc,
                            const float* __restrict__ Ds,
                            _Float16* __restrict__ g16) {
    int idx = blockIdx.x * blockDim.x + threadIdx.x;
    if (idx >= M_TOK * D_INNER) return;
    int token = idx / D_INNER;
    int d     = idx % D_INNER;
    size_t xb = (size_t)token * (2 * D_INNER);
    float y = xz[xb + d];
    float z = xz[xb + D_INNER + d];
    float g = (y + uc[idx] * Ds[d]) * silu_f(z);
    g16[idx] = (_Float16)g;
}

// ---------------- final RMSNorm of cls token ----------------
__global__ void final_rms_kernel(const float* __restrict__ x,
                                 const float* __restrict__ resid,
                                 const float* __restrict__ w,
                                 float* __restrict__ xcls) {
    int wv   = threadIdx.x >> 5;
    int lane = threadIdx.x & 31;
    int b = blockIdx.x * (blockDim.x >> 5) + wv;
    if (b >= BATCH) return;
    size_t base = (size_t)(b * L_SEQ + N_PATCH) * D_MODEL;
    float vals[6];
    float ss = 0.0f;
    #pragma unroll
    for (int i = 0; i < 6; ++i) {
        int c = lane + 32 * i;
        float v = x[base + c] + resid[base + c];
        vals[i] = v;
        ss += v * v;
    }
    #pragma unroll
    for (int off = 16; off > 0; off >>= 1)
        ss += __shfl_xor(ss, off);
    float inv = rsqrtf(ss * (1.0f / D_MODEL) + EPSF);
    #pragma unroll
    for (int i = 0; i < 6; ++i) {
        int c = lane + 32 * i;
        xcls[b * D_MODEL + c] = vals[i] * inv * w[c];
    }
}

// ---------------- head ----------------
__global__ void head_kernel(const float* __restrict__ xcls,
                            const float* __restrict__ head_w,
                            const float* __restrict__ head_b,
                            float* __restrict__ out) {
    int idx = blockIdx.x * blockDim.x + threadIdx.x;
    if (idx >= BATCH * NCLS) return;
    int b = idx / NCLS;
    int n = idx % NCLS;
    const float* xr = xcls + b * D_MODEL;
    float s = head_b[n];
    for (int c = 0; c < D_MODEL; ++c)
        s += xr[c] * head_w[c * NCLS + n];
    out[idx] = s;
}

// ---------------- launcher ----------------
extern "C" void kernel_launch(void* const* d_in, const int* in_sizes, int n_in,
                              void* d_out, int out_size, void* d_ws, size_t ws_size,
                              hipStream_t stream) {
    const float* imgs       = (const float*)d_in[0];
    const float* patch_w    = (const float*)d_in[1];
    const float* patch_b    = (const float*)d_in[2];
    const float* cls_token  = (const float*)d_in[3];
    const float* pos_embed  = (const float*)d_in[4];
    const float* norm_ws    = (const float*)d_in[5];
    const float* in_proj_ws = (const float*)d_in[6];
    const float* conv_ws    = (const float*)d_in[7];
    const float* conv_bs    = (const float*)d_in[8];
    const float* x_proj_ws  = (const float*)d_in[9];
    const float* dt_proj_ws = (const float*)d_in[10];
    const float* dt_proj_bs = (const float*)d_in[11];
    const float* A_logs     = (const float*)d_in[12];
    const float* Ds         = (const float*)d_in[13];
    const float* out_proj_ws= (const float*)d_in[14];
    const float* norm_f_w   = (const float*)d_in[15];
    const float* head_w     = (const float*)d_in[16];
    const float* head_b     = (const float*)d_in[17];
    float* out = (float*)d_out;

    // workspace carve-up (all fully rewritten every call)
    char* ws = (char*)d_ws;
    size_t off = 0;
    auto carve = [&](size_t bytes) -> char* {
        char* p = ws + off;
        off = (off + bytes + 255) & ~(size_t)255;
        return p;
    };
    float*    x      = (float*)   carve((size_t)M_TOK * D_MODEL * 4);
    float*    resid  = (float*)   carve((size_t)M_TOK * D_MODEL * 4);
    float*    xz     = (float*)   carve((size_t)M_TOK * 2 * D_INNER * 4);
    float*    uc     = (float*)   carve((size_t)M_TOK * D_INNER * 4);
    float*    delta  = (float*)   carve((size_t)M_TOK * D_INNER * 4);
    float*    dbl    = (float*)   carve((size_t)M_TOK * (DT_RANK + 2 * D_STATE) * 4);
    _Float16* a16    = (_Float16*)carve((size_t)M_TOK * D_INNER * 2);
    _Float16* w16    = (_Float16*)carve((size_t)D_MODEL * 2 * D_INNER * 2);
    float*    xcls   = (float*)   carve((size_t)BATCH * D_MODEL * 4);
    (void)ws_size;

    const int TB = 256;
    auto cdiv = [](int a, int b) { return (a + b - 1) / b; };

    // patch + pos embed
    patch_embed_kernel<<<cdiv(M_TOK * D_MODEL, TB), TB, 0, stream>>>(
        imgs, patch_w, patch_b, cls_token, pos_embed, x);

    for (int layer = 0; layer < DEPTH; ++layer) {
        const float* nw   = norm_ws     + layer * D_MODEL;
        const float* ipw  = in_proj_ws  + (size_t)layer * D_MODEL * 2 * D_INNER;
        const float* cw   = conv_ws     + (size_t)layer * D_INNER * D_CONV;
        const float* cb   = conv_bs     + (size_t)layer * D_INNER;
        const float* xpw  = x_proj_ws   + (size_t)layer * D_INNER * (DT_RANK + 2 * D_STATE);
        const float* dtw  = dt_proj_ws  + (size_t)layer * DT_RANK * D_INNER;
        const float* dtb  = dt_proj_bs  + (size_t)layer * D_INNER;
        const float* alog = A_logs      + (size_t)layer * D_INNER * D_STATE;
        const float* dp   = Ds          + (size_t)layer * D_INNER;
        const float* opw  = out_proj_ws + (size_t)layer * D_INNER * D_MODEL;

        // residual + rmsnorm -> a16 (K = 192)
        rmsnorm_kernel<<<cdiv(M_TOK, 8), TB, 0, stream>>>(
            x, resid, a16, nw, layer == 0 ? 1 : 0);

        // in_proj: [12832 x 192] x [192 x 768] -> xz
        cvt_f16_kernel<<<cdiv(D_MODEL * 2 * D_INNER, TB), TB, 0, stream>>>(
            ipw, w16, D_MODEL * 2 * D_INNER);
        {
            dim3 grid((2 * D_INNER) / 64, cdiv(M_TOK, 64));
            gemm_f16_wmma_kernel<<<grid, 128, 0, stream>>>(
                a16, w16, xz, M_TOK, 2 * D_INNER, D_MODEL);
        }

        conv_silu_kernel<<<cdiv(M_TOK * D_INNER, TB), TB, 0, stream>>>(xz, cw, cb, uc);
        xproj_kernel<<<cdiv(M_TOK * (DT_RANK + 2 * D_STATE), TB), TB, 0, stream>>>(uc, xpw, dbl);
        dtproj_kernel<<<cdiv(M_TOK * D_INNER, TB), TB, 0, stream>>>(dbl, dtw, dtb, delta);
        scan_kernel<<<cdiv(BATCH * D_INNER, TB), TB, 0, stream>>>(delta, uc, dbl, alog, xz);
        gate_kernel<<<cdiv(M_TOK * D_INNER, TB), TB, 0, stream>>>(xz, uc, dp, a16);

        // out_proj: [12832 x 384] x [384 x 192] -> x
        cvt_f16_kernel<<<cdiv(D_INNER * D_MODEL, TB), TB, 0, stream>>>(
            opw, w16, D_INNER * D_MODEL);
        {
            dim3 grid(D_MODEL / 64, cdiv(M_TOK, 64));
            gemm_f16_wmma_kernel<<<grid, 128, 0, stream>>>(
                a16, w16, x, M_TOK, D_MODEL, D_INNER);
        }
    }

    final_rms_kernel<<<cdiv(BATCH, 8), TB, 0, stream>>>(x, resid, norm_f_w, xcls);
    head_kernel<<<cdiv(BATCH * NCLS, TB), TB, 0, stream>>>(xcls, head_w, head_b, out);
    (void)in_sizes; (void)n_in; (void)out_size;
}
